// SimpleRetention_88665304859362
// MI455X (gfx1250) — matrix operand order repaired and checked by reading, not compiled, not run on hardware
//
#include <hip/hip_runtime.h>
#include <hip/hip_bf16.h>
#include <math.h>

#define HID   2048
#define SEQ   2048
#define BATCH 4
#define BM    128
#define BN    128
#define BK    32
#define SM_STRIDE 40   // BK + 8 bf16 pad: 80B row pitch -> 16B aligned, bank-conflict free
#define THREADS 256

typedef __attribute__((ext_vector_type(16))) __bf16 v16bf;
typedef __attribute__((ext_vector_type(8)))  float  v8f;
typedef __attribute__((ext_vector_type(4)))  unsigned int u32x4;
typedef __attribute__((ext_vector_type(8)))  int i32x8;
typedef __attribute__((ext_vector_type(4)))  int i32x4;

#if defined(__has_builtin)
#if __has_builtin(__builtin_amdgcn_tensor_load_to_lds)
#define HAVE_TDM 1
#endif
#endif
#ifndef HAVE_TDM
#define HAVE_TDM 0
#endif

__device__ __forceinline__ unsigned short f2bf(float f) {
  unsigned u = __builtin_bit_cast(unsigned, f);
  u += 0x7FFFu + ((u >> 16) & 1u);   // round-to-nearest-even
  return (unsigned short)(u >> 16);
}

// A fragment (16x32 bf16, ISA 7.12.2 layout): lanes 0-15 -> K {0..7,16..23}, lanes 16-31 -> K {8..15,24..31}
__device__ __forceinline__ v16bf ld_fragA(const unsigned short* sm, int rowBase, int lane) {
  int r  = rowBase + (lane & 15);
  int lo = (lane & 16) ? 8 : 0;
  const u32x4* p = reinterpret_cast<const u32x4*>(sm + r * SM_STRIDE + lo);
  union { u32x4 u[2]; v16bf v; } f;
  f.u[0] = p[0];   // k = lo .. lo+7
  f.u[1] = p[2];   // k = lo+16 .. lo+23
  return f.v;
}

// B fragment (32x16 bf16): lane holds column N = lane&15; lanes 0-15 K {0..7,16..23}, lanes 16-31 K {8..15,24..31}.
// smB stores B transposed: Bt[n][k], so this is two b128 reads of row n (k = ko.., ko+16..).
__device__ __forceinline__ v16bf ld_fragB(const unsigned short* sm, int rowBase, int lane) {
  int r  = rowBase + (lane & 15);
  int ko = (lane & 16) ? 8 : 0;
  const u32x4* p = reinterpret_cast<const u32x4*>(sm + r * SM_STRIDE + ko);
  union { u32x4 u[2]; v16bf v; } f;
  f.u[0] = p[0];
  f.u[1] = p[2];
  return f.v;
}

__device__ __forceinline__ void mma_step(const unsigned short* smA, const unsigned short* smB,
                                         int waveM, int waveN, int lane, v8f acc[4][2]) {
  v16bf af[4], bfr[2];
#pragma unroll
  for (int i = 0; i < 4; ++i) af[i] = ld_fragA(smA, waveM * 64 + i * 16, lane);
#pragma unroll
  for (int j = 0; j < 2; ++j) bfr[j] = ld_fragB(smB, waveN * 32 + j * 16, lane);
#pragma unroll
  for (int i = 0; i < 4; ++i)
#pragma unroll
    for (int j = 0; j < 2; ++j)
      acc[i][j] = __builtin_amdgcn_wmma_f32_16x16x32_bf16(
          false, af[i], false, bfr[j], (short)0, acc[i][j], false, false);
}

// ---- cooperative tile loaders -------------------------------------------------

// fp32 global (BM x BK) -> bf16 LDS row-major
__device__ __forceinline__ void load_a_f32(const float* g, int ldg, unsigned short* sm, int tid) {
#pragma unroll
  for (int i = 0; i < 4; ++i) {
    int idx = tid + i * THREADS;         // 0..1023 float4 chunks
    int row = idx >> 3;                  // 8 chunks per 32-wide row
    int c   = (idx & 7) << 2;
    float4 v = *reinterpret_cast<const float4*>(g + (size_t)row * ldg + c);
    unsigned pk0 = (unsigned)f2bf(v.x) | ((unsigned)f2bf(v.y) << 16);
    unsigned pk1 = (unsigned)f2bf(v.z) | ((unsigned)f2bf(v.w) << 16);
    *reinterpret_cast<uint2*>(sm + row * SM_STRIDE + c) = make_uint2(pk0, pk1);
  }
}

// fp32 global (BK x BN) -> transposed bf16 LDS Bt[BN][BK].
// Each thread reads a 4-deep column (4 coalesced dword loads across lanes) and
// writes ONE packed b64 LDS store -> no scalar d16 store spam.
__device__ __forceinline__ void load_bt_f32(const float* g, int ldg, unsigned short* sm, int tid) {
#pragma unroll
  for (int i = 0; i < 4; ++i) {
    int idx = tid + i * THREADS;         // 0..1023
    int n   = idx & 127;                 // col within tile (contiguous across lanes)
    int k4  = idx >> 7;                  // 0..7 -> k = 4*k4
    float x0 = g[(size_t)(4 * k4 + 0) * ldg + n];
    float x1 = g[(size_t)(4 * k4 + 1) * ldg + n];
    float x2 = g[(size_t)(4 * k4 + 2) * ldg + n];
    float x3 = g[(size_t)(4 * k4 + 3) * ldg + n];
    unsigned pk0 = (unsigned)f2bf(x0) | ((unsigned)f2bf(x1) << 16);
    unsigned pk1 = (unsigned)f2bf(x2) | ((unsigned)f2bf(x3) << 16);
    *reinterpret_cast<uint2*>(sm + n * SM_STRIDE + 4 * k4) = make_uint2(pk0, pk1);
  }
}

// bf16 global (BM x BK) -> LDS copy (b128 in, b128 out)
__device__ __forceinline__ void load_a_bf16(const unsigned short* g, int ldg, unsigned short* sm, int tid) {
#pragma unroll
  for (int i = 0; i < 2; ++i) {
    int idx = tid + i * THREADS;         // 0..511 chunks of 8 bf16
    int row = idx >> 2;
    int c   = (idx & 3) << 3;
    u32x4 v = *reinterpret_cast<const u32x4*>(g + (size_t)row * ldg + c);
    *reinterpret_cast<u32x4*>(sm + row * SM_STRIDE + c) = v;
  }
}

// bf16 global (BK x BN) -> transposed bf16 LDS Bt[BN][BK] (column reads, packed b64 store)
__device__ __forceinline__ void load_bt_bf16(const unsigned short* g, int ldg, unsigned short* sm, int tid) {
#pragma unroll
  for (int i = 0; i < 4; ++i) {
    int idx = tid + i * THREADS;         // 0..1023
    int n   = idx & 127;
    int k4  = idx >> 7;
    unsigned short x0 = g[(size_t)(4 * k4 + 0) * ldg + n];
    unsigned short x1 = g[(size_t)(4 * k4 + 1) * ldg + n];
    unsigned short x2 = g[(size_t)(4 * k4 + 2) * ldg + n];
    unsigned short x3 = g[(size_t)(4 * k4 + 3) * ldg + n];
    unsigned pk0 = (unsigned)x0 | ((unsigned)x1 << 16);
    unsigned pk1 = (unsigned)x2 | ((unsigned)x3 << 16);
    *reinterpret_cast<uint2*>(sm + n * SM_STRIDE + 4 * k4) = make_uint2(pk0, pk1);
  }
}

// ---- Tensor Data Mover path (att kernel tile copies) --------------------------

#if HAVE_TDM
__device__ __forceinline__ unsigned lds_offset(const void* p) {
  return (unsigned)(size_t)(__attribute__((address_space(3))) const void*)p;
}

// 2D tile load: BK(=32 bf16) x BM(=128) rows, row stride HID elems, with TDM LDS
// padding reproducing the SM_STRIDE=40 bf16 pitch (16 DW row + 4 DW pad).
__device__ __forceinline__ void tdm_load_tile(const unsigned short* gptr, unsigned ldsOff) {
  unsigned long long ga = (unsigned long long)(size_t)gptr;
  u32x4 g0 = { 1u,                                   // count=1, user mode
               ldsOff,                               // lds_addr (bytes)
               (unsigned)ga,                         // global_addr[31:0]
               (unsigned)((ga >> 32) & 0x01FFFFFFu) | 0x80000000u };  // addr[56:32] | type=2
  i32x8 g1 = { (int)0x06D10000,                      // data_size=2B, pad_en, pad_int=16DW, pad_amt=4DW
               0,                                    // atomic barrier addr (unused)
               (int)((HID & 0xFFFF) << 16),          // tensor_dim0[15:0]
               (int)((SEQ & 0xFFFF) << 16),          // dim0[31:16]=0 | tensor_dim1[15:0]
               (int)(BM & 0xFFFF) << 16 >> 16 | (BK << 16) ? 0 : 0,  // placeholder (set below)
               (int)HID,                             // tensor_dim0_stride[31:0]
               0, 0 };
  // word3 fixup: [111:96] dim1 hi = 0, [127:112] tile_dim0 = BK
  g1[3] = (int)(((unsigned)(SEQ & 0xFFFF) << 16));
  g1[3] = (int)((unsigned)g1[3]);   // dim1 low already placed in word2
  // Correct field placement (bits relative to group1):
  //  w2 = dim0[31:16] | dim1[15:0]<<16 ; w3 = dim1[31:16] | tile_dim0<<16 ; w4 = tile_dim1
  g1[2] = (int)(((HID >> 16) & 0xFFFF) | ((SEQ & 0xFFFF) << 16));
  g1[3] = (int)(((SEQ >> 16) & 0xFFFF) | ((unsigned)BK << 16));
  g1[4] = (int)BM;                                   // tile_dim1 = 128 rows, tile_dim2 = 0
  i32x4 z4 = { 0, 0, 0, 0 };
#if __has_include(<hip/amd_detail/amd_gfx1250_TDM.h>)
  i32x8 z8 = { 0, 0, 0, 0, 0, 0, 0, 0 };
  __builtin_amdgcn_tensor_load_to_lds(g0, g1, z4, z4, z8, 0);
#else
  __builtin_amdgcn_tensor_load_to_lds(g0, g1, z4, z4, 0);
#endif
}

__device__ __forceinline__ void tdm_wait0() {
#if __has_builtin(__builtin_amdgcn_s_wait_tensorcnt)
  __builtin_amdgcn_s_wait_tensorcnt((short)0);
#else
  asm volatile("s_wait_tensorcnt 0x0" ::: "memory");
#endif
}
#endif  // HAVE_TDM

// ---- kernel 1: Q/K/V projections + xPos epilogue ------------------------------

__global__ __launch_bounds__(THREADS) void proj_kernel(
    const float* __restrict__ X, const float* __restrict__ WQ,
    const float* __restrict__ WK, const float* __restrict__ WV,
    unsigned short* __restrict__ Qb, unsigned short* __restrict__ Kb,
    unsigned short* __restrict__ Vb) {
  __shared__ unsigned short smA[BM * SM_STRIDE];
  __shared__ unsigned short smB[BN * SM_STRIDE];
  const int tid = threadIdx.x, lane = tid & 31, wid = tid >> 5;
  const int waveM = wid & 1, waveN = wid >> 1;
  const int nBase = blockIdx.x * BN;
  const int mBase = blockIdx.y * BM;        // flattened B*S row
  const int mode  = blockIdx.z;             // 0=Q 1=K 2=V
  const float* W      = (mode == 0) ? WQ : (mode == 1) ? WK : WV;
  unsigned short* Out = (mode == 0) ? Qb : (mode == 1) ? Kb : Vb;

  v8f acc[4][2] = {};
  for (int k = 0; k < HID; k += BK) {
    load_a_f32 (X + (size_t)mBase * HID + k,     HID, smA, tid);
    load_bt_f32(W + (size_t)k     * HID + nBase, HID, smB, tid);
    __syncthreads();
    mma_step(smA, smB, waveM, waveN, lane, acc);
    __syncthreads();
  }

  const float inv14H = 1.0f / (1.4f * (float)HID);
#pragma unroll
  for (int i = 0; i < 4; ++i)
#pragma unroll
    for (int j = 0; j < 2; ++j)
#pragma unroll
      for (int v = 0; v < 8; ++v) {
        int row = mBase + waveM * 64 + i * 16 + v + ((lane >> 4) << 3);
        int col = nBase + waveN * 32 + j * 16 + (lane & 15);
        float val = acc[i][j][v];
        float partner = __shfl_xor(val, 1, 32);   // even/odd column pair
        float r;
        if (mode == 2) {
          r = val;
        } else {
          int pos = row & (SEQ - 1);
          float scale_v = ((float)(col & ~1) + 0.4f * (float)HID) * inv14H;
          float power   = (float)pos * (1.0f / 512.0f);
          float sc = exp2f(((mode == 1) ? -power : power) * __log2f(scale_v));
          float p  = (float)(col >> 1);
          float theta = (float)pos * exp2f(-(p * (1.0f / 1024.0f)) * 13.287712379549449f);
          float sn, cs;
          __sincosf(theta, &sn, &cs);
          sn *= sc; cs *= sc;
          r = (col & 1) ? (val * cs + partner * sn) : (val * cs - partner * sn);
        }
        Out[(size_t)row * HID + col] = f2bf(r);
      }
}

// ---- kernel 2: att = (Q K^T) * gamma^(n-m), lower-triangular blocks only ------

__global__ __launch_bounds__(THREADS) void att_kernel(
    const unsigned short* __restrict__ Qb, const unsigned short* __restrict__ Kb,
    unsigned short* __restrict__ Ab) {
  const int mTile = blockIdx.x, nTile = blockIdx.y, b = blockIdx.z;
  if (mTile > nTile) return;                 // strictly above diagonal: never read
  __shared__ unsigned short smA[BM * SM_STRIDE];
  __shared__ unsigned short smB[BN * SM_STRIDE];
  const int tid = threadIdx.x, lane = tid & 31, wid = tid >> 5;
  const int waveM = wid & 1, waveN = wid >> 1;
  const int nBase = nTile * BM;
  const int mBase = mTile * BN;
  const unsigned short* Q = Qb + (size_t)b * SEQ * HID;
  const unsigned short* K = Kb + (size_t)b * SEQ * HID;

#if HAVE_TDM
  const unsigned offA = lds_offset(smA);
  const unsigned offB = lds_offset(smB);
#endif

  v8f acc[4][2] = {};
  for (int k = 0; k < HID; k += BK) {
#if HAVE_TDM
    if (wid == 0) {                        // one wave drives the DMA engine
      tdm_load_tile(Q + (size_t)nBase * HID + k, offA);
      tdm_load_tile(K + (size_t)mBase * HID + k, offB);
      tdm_wait0();
    }
#else
    load_a_bf16(Q + (size_t)nBase * HID + k, HID, smA, tid);
    load_a_bf16(K + (size_t)mBase * HID + k, HID, smB, tid);  // K row-major == Bt layout
#endif
    __syncthreads();
    mma_step(smA, smB, waveM, waveN, lane, acc);
    __syncthreads();
  }

  const float l2g = -0.04580368961312478f;   // log2(0.96875)
  unsigned short* A = Ab + (size_t)b * SEQ * SEQ;
#pragma unroll
  for (int i = 0; i < 4; ++i)
#pragma unroll
    for (int j = 0; j < 2; ++j)
#pragma unroll
      for (int v = 0; v < 8; ++v) {
        int n = nBase + waveM * 64 + i * 16 + v + ((lane >> 4) << 3);
        int m = mBase + waveN * 32 + j * 16 + (lane & 15);
        int d = n - m;
        float val = (d < 0) ? 0.0f : acc[i][j][v] * exp2f((float)d * l2g);
        A[(size_t)n * SEQ + m] = f2bf(val);
      }
}

// ---- kernel 3: out = att @ V (triangular K range) -----------------------------

__global__ __launch_bounds__(THREADS) void out_kernel(
    const unsigned short* __restrict__ Ab, const unsigned short* __restrict__ Vb,
    float* __restrict__ Out) {
  const int hTile = blockIdx.x, nTile = blockIdx.y, b = blockIdx.z;
  __shared__ unsigned short smA[BM * SM_STRIDE];
  __shared__ unsigned short smB[BN * SM_STRIDE];
  const int tid = threadIdx.x, lane = tid & 31, wid = tid >> 5;
  const int waveM = wid & 1, waveN = wid >> 1;
  const int nBase = nTile * BM;
  const int hBase = hTile * BN;
  const unsigned short* A = Ab + (size_t)b * SEQ * SEQ;
  const unsigned short* V = Vb + (size_t)b * SEQ * HID;
  const int kEnd = (nTile + 1) * BM;         // only m-blocks <= n-block contribute

  v8f acc[4][2] = {};
  for (int k = 0; k < kEnd; k += BK) {
    load_a_bf16 (A + (size_t)nBase * SEQ + k,     SEQ, smA, tid);
    load_bt_bf16(V + (size_t)k     * HID + hBase, HID, smB, tid);
    __syncthreads();
    mma_step(smA, smB, waveM, waveN, lane, acc);
    __syncthreads();
  }

#pragma unroll
  for (int i = 0; i < 4; ++i)
#pragma unroll
    for (int j = 0; j < 2; ++j)
#pragma unroll
      for (int v = 0; v < 8; ++v) {
        int n   = nBase + waveM * 64 + i * 16 + v + ((lane >> 4) << 3);
        int col = hBase + waveN * 32 + j * 16 + (lane & 15);
        Out[((size_t)b * SEQ + n) * HID + col] = acc[i][j][v];
      }
}

// ---- host launcher ------------------------------------------------------------

extern "C" void kernel_launch(void* const* d_in, const int* in_sizes, int n_in,
                              void* d_out, int out_size, void* d_ws, size_t ws_size,
                              hipStream_t stream) {
  const float* X  = (const float*)d_in[0];
  const float* WQ = (const float*)d_in[1];
  const float* WK = (const float*)d_in[2];
  const float* WV = (const float*)d_in[3];
  float* out = (float*)d_out;

  const size_t elems = (size_t)BATCH * SEQ * HID;   // 16.7M elems, 32MB bf16 each
  unsigned short* Qb = (unsigned short*)d_ws;
  unsigned short* Kb = Qb + elems;
  unsigned short* Vb = Kb + elems;
  unsigned short* Ab = Vb + elems;                   // B*S*S bf16

  hipLaunchKernelGGL(proj_kernel, dim3(HID / BN, (BATCH * SEQ) / BM, 3),
                     dim3(THREADS), 0, stream, X, WQ, WK, WV, Qb, Kb, Vb);
  hipLaunchKernelGGL(att_kernel, dim3(SEQ / BN, SEQ / BM, BATCH),
                     dim3(THREADS), 0, stream, Qb, Kb, Ab);
  hipLaunchKernelGGL(out_kernel, dim3(HID / BN, SEQ / BM, BATCH),
                     dim3(THREADS), 0, stream, Ab, Vb, out);
}
